// LMUCell_57913339019730
// MI455X (gfx1250) — matrix operand order repaired and checked by reading, not compiled, not run on hardware
//
#include <hip/hip_runtime.h>
#include <hip/hip_bf16.h>
#include <math.h>

// ---------------------------------------------------------------------------
// LMU recurrent scan for MI455X (gfx1250, wave32).
// B=32, T=2048, D=128, U=512, N=256.
// Persistent kernel, 16 workgroups, one grid barrier per time step.
//   m-recurrence : v_wmma_f32_16x16x4_f32   (fp32, precision critical)
//   h-update     : v_wmma_f32_16x16x32_bf16 (bf16 inputs, f32 accumulate)
// Per-WG weight slices (Wi/Wh/Wm fragments, 56 KB) are cached in LDS once;
// only the AT fragment stream (full 256 KB, L2-resident) stays in global.
// ---------------------------------------------------------------------------

#define LMU_B 32
#define LMU_T 2048
#define LMU_D 128
#define LMU_U 512
#define LMU_N 256
#define LMU_G 16   // persistent workgroups

typedef __attribute__((ext_vector_type(2)))  float  v2f;
typedef __attribute__((ext_vector_type(8)))  float  v8f;
typedef __attribute__((ext_vector_type(16))) __bf16 v16bf;

// ---------------- WMMA wrappers -------------------------------------------
__device__ __forceinline__ v8f wmma_bf16(v16bf a, v16bf b, v8f c) {
  return __builtin_amdgcn_wmma_f32_16x16x32_bf16(false, a, false, b,
                                                 (short)0, c, false, false);
}
__device__ __forceinline__ v8f wmma_f32(v2f a, v2f b, v8f c) {
  return __builtin_amdgcn_wmma_f32_16x16x4_f32(false, a, false, b,
                                               (short)0, c, false, false);
}

// A-fragment (16x32 bf16) built from an fp32 row.
// lane: m = lane&15, half = lane>>4. elem i -> k = 16*half+i.
// p points at row + 32*kt + 16*half.
__device__ __forceinline__ v16bf a_from_f32(const float* p) {
  v16bf a;
#pragma unroll
  for (int i = 0; i < 8; ++i) {
    a[i]     = (__bf16)p[i];
    a[i + 8] = (__bf16)p[8 + i];
  }
  return a;
}
__device__ __forceinline__ v16bf a_from_bf16(const __bf16* p) {
  v16bf a;
#pragma unroll
  for (int i = 0; i < 8; ++i) {
    a[i]     = p[i];
    a[i + 8] = p[8 + i];
  }
  return a;
}

// ---------------- prep: pack weights into WMMA fragment layout -------------
// bf16 B-fragment (32x16), lane L: n = 16*nt + (L&15), k = 32*kt + 16*(L>>4)+i
__global__ void lmu_pack_w(const float* __restrict__ w, __bf16* __restrict__ wf,
                           int K) {
  int e = blockIdx.x * blockDim.x + threadIdx.x;
  if (e >= K * LMU_U) return;
  int frag = e >> 9;           // 512 elems per fragment
  int lane = (e >> 4) & 31;
  int i    = e & 15;
  int kt   = frag >> 5;        // NT = 32 n-tiles (U=512)
  int nt   = frag & 31;
  int k    = 32 * kt + 16 * (lane >> 4) + i;
  int n    = 16 * nt + (lane & 15);
  wf[e] = (__bf16)w[(size_t)k * LMU_U + n];
}

// f32 B-fragment (4x16) for 16x16x4: lane L: n = 16*nt+(L&15), k = 4*kt+2*(L>>4)+v
__global__ void lmu_pack_at(const float* __restrict__ at, float* __restrict__ atf) {
  int e = blockIdx.x * blockDim.x + threadIdx.x;   // 64*16*32*2 = 65536
  if (e >= 64 * 16 * 64) return;
  int frag = e >> 6;
  int lane = (e >> 1) & 31;
  int v    = e & 1;
  int kt   = frag >> 4;
  int nt   = frag & 15;
  int k    = 4 * kt + 2 * (lane >> 4) + v;
  int n    = 16 * nt + (lane & 15);
  atf[e] = at[k * LMU_N + n];
}

// ---------------- persistent scan kernel -----------------------------------
__global__ void __launch_bounds__(256)
lmu_scan(const float* __restrict__ x,   const float* __restrict__ h0,
         const float* __restrict__ m0,  const float* __restrict__ ie,
         const float* __restrict__ he,  const float* __restrict__ me,
         const float* __restrict__ bt,  const float* __restrict__ atf,
         const __bf16* __restrict__ wif, const __bf16* __restrict__ whf,
         const __bf16* __restrict__ wmf, float* __restrict__ out,
         unsigned* __restrict__ counter) {
  // LDS: fp32 m state (padded rows), bf16 mirror, u, and the per-WG
  // weight-fragment cache (28 k-tiles x 2 n-tiles x 1KB = 56KB).
  __shared__ __align__(16) float  m_lds[LMU_B][LMU_N + 2];
  __shared__ __align__(16) __bf16 mbf[LMU_B][LMU_N + 8];
  __shared__ float u_lds[LMU_B];
  __shared__ __align__(32) __bf16 wlds[56 * 512];

  const int tid  = threadIdx.x;
  const int w    = tid >> 5;
  const int lane = tid & 31;
  const int half = lane >> 4;
  const int lnn  = lane & 15;
  const int wg   = blockIdx.x;

  // ---- one-time: init m state = m0, preload per-WG weight fragments -------
  for (int idx = tid; idx < LMU_B * LMU_N; idx += 256)
    m_lds[idx / LMU_N][idx % LMU_N] = m0[idx];
  for (int idx = tid; idx < 56 * 512; idx += 256) {
    int f = idx >> 9, e = idx & 511;
    const __bf16* src;
    int kt, j;
    if (f < 8)       { src = wif; kt = f >> 1;        j = f & 1; }
    else if (f < 40) { src = whf; kt = (f - 8) >> 1;  j = f & 1; }
    else             { src = wmf; kt = (f - 40) >> 1; j = f & 1; }
    int g = kt * 32 + (wg * 2 + j);        // global fragment id in packed array
    wlds[idx] = src[(size_t)g * 512 + e];
  }
  __syncthreads();

  // per-wave constants for the m-update (hoisted out of the time loop)
  float btn[2];
#pragma unroll
  for (int j = 0; j < 2; ++j) btn[j] = bt[(2 * w + j) * 16 + lnn];

  for (int t = 0; t < LMU_T; ++t) {
    // ---- phase 1: u[b] = x_t@ie + h(t-1)@he + m@me  (deterministic GEMV) ---
    {
      for (int r = 0; r < 4; ++r) {
        int bb = w * 4 + r;
        float s = 0.f;
        const float* xr = x + ((size_t)bb * LMU_T + t) * LMU_D;
        for (int i = lane; i < LMU_D; i += 32) s += xr[i] * ie[i];
        const float* hr = (t == 0)
            ? h0 + (size_t)bb * LMU_U
            : out + (size_t)bb * LMU_T * LMU_U + (size_t)(t - 1) * LMU_U;
        for (int i = lane; i < LMU_U; i += 32) s += hr[i] * he[i];
        for (int i = lane; i < LMU_N; i += 32) s += m_lds[bb][i] * me[i];
#pragma unroll
        for (int off = 16; off > 0; off >>= 1) s += __shfl_down(s, off, 32);
        if (lane == 0) u_lds[bb] = s;
      }
    }
    __syncthreads();

    // ---- phase 2: m_new = m + m@AT + u (x) BT   (fp32 WMMA 16x16x4) -------
    v8f acc[4];
    {
#pragma unroll
      for (int mt = 0; mt < 2; ++mt)
#pragma unroll
        for (int j = 0; j < 2; ++j) {
          int n = (2 * w + j) * 16 + lnn;
          v8f c;
#pragma unroll
          for (int v = 0; v < 8; ++v) {
            int bb = mt * 16 + v + 8 * half;
            c[v] = m_lds[bb][n] + u_lds[bb] * btn[j];
          }
          acc[mt * 2 + j] = c;
        }
      for (int kt = 0; kt < 64; ++kt) {
        v2f a0 = *(const v2f*)&m_lds[lnn][4 * kt + 2 * half];
        v2f a1 = *(const v2f*)&m_lds[16 + lnn][4 * kt + 2 * half];
#pragma unroll
        for (int j = 0; j < 2; ++j) {
          int ntt = 2 * w + j;
          v2f bv = *(const v2f*)(atf + ((size_t)(kt * 16 + ntt) * 32 + lane) * 2);
          acc[0 * 2 + j] = wmma_f32(a0, bv, acc[0 * 2 + j]);
          acc[1 * 2 + j] = wmma_f32(a1, bv, acc[1 * 2 + j]);
        }
      }
    }
    __syncthreads();   // all reads of m_old done before overwrite

    // ---- phase 3: write back m_new (fp32 + bf16 mirror) --------------------
#pragma unroll
    for (int mt = 0; mt < 2; ++mt)
#pragma unroll
      for (int j = 0; j < 2; ++j) {
        int n = (2 * w + j) * 16 + lnn;
        v8f c = acc[mt * 2 + j];
#pragma unroll
        for (int v = 0; v < 8; ++v) {
          int bb = mt * 16 + v + 8 * half;
          m_lds[bb][n] = c[v];
          mbf[bb][n]   = (__bf16)c[v];
        }
      }
    __syncthreads();

    // ---- phase 4: h = tanh(x@Wi + h@Wh + m_new@Wm)  (bf16 WMMA, sliced) ----
    // B-fragments come from the LDS weight cache: frag = base + kt*2 + ntl.
    if (w < 4) {
      int mt  = w >> 1, ntl = w & 1;
      int ntg = wg * 2 + ntl;                 // global n-tile 0..31
      int row = mt * 16 + lnn;
      v8f hacc;
#pragma unroll
      for (int v = 0; v < 8; ++v) hacc[v] = 0.f;

      const float* xr = x + ((size_t)row * LMU_T + t) * LMU_D;
#pragma unroll
      for (int kt = 0; kt < 4; ++kt) {        // Wi, K = 128
        v16bf a = a_from_f32(xr + 32 * kt + 16 * half);
        const v16bf* bp =
            (const v16bf*)&wlds[(size_t)(0 + kt * 2 + ntl) * 512 + lane * 16];
        hacc = wmma_bf16(a, *bp, hacc);
      }
      const float* hr = (t == 0)
          ? h0 + (size_t)row * LMU_U
          : out + (size_t)row * LMU_T * LMU_U + (size_t)(t - 1) * LMU_U;
      for (int kt = 0; kt < 16; ++kt) {       // Wh, K = 512
        v16bf a = a_from_f32(hr + 32 * kt + 16 * half);
        const v16bf* bp =
            (const v16bf*)&wlds[(size_t)(8 + kt * 2 + ntl) * 512 + lane * 16];
        hacc = wmma_bf16(a, *bp, hacc);
      }
      const __bf16* mr = &mbf[row][0];
#pragma unroll
      for (int kt = 0; kt < 8; ++kt) {        // Wm, K = 256 (LDS)
        v16bf a = a_from_bf16(mr + 32 * kt + 16 * half);
        const v16bf* bp =
            (const v16bf*)&wlds[(size_t)(40 + kt * 2 + ntl) * 512 + lane * 16];
        hacc = wmma_bf16(a, *bp, hacc);
      }
      int n = ntg * 16 + lnn;
#pragma unroll
      for (int v = 0; v < 8; ++v) {
        int bb = mt * 16 + v + 8 * half;
        out[(size_t)bb * LMU_T * LMU_U + (size_t)t * LMU_U + n] = tanhf(hacc[v]);
      }
    }

    // ---- grid-wide barrier (release h(t), acquire for next step) ----------
    __threadfence();
    __syncthreads();
    if (tid == 0) {
      __hip_atomic_fetch_add(counter, 1u, __ATOMIC_RELEASE,
                             __HIP_MEMORY_SCOPE_AGENT);
      unsigned target = (unsigned)LMU_G * (unsigned)(t + 1);
      while (__hip_atomic_load(counter, __ATOMIC_ACQUIRE,
                               __HIP_MEMORY_SCOPE_AGENT) < target)
        __builtin_amdgcn_s_sleep(2);
    }
    __syncthreads();
  }
}

// ---------------------------------------------------------------------------
extern "C" void kernel_launch(void* const* d_in, const int* in_sizes, int n_in,
                              void* d_out, int out_size, void* d_ws,
                              size_t ws_size, hipStream_t stream) {
  const float* x  = (const float*)d_in[0];
  const float* h0 = (const float*)d_in[1];
  const float* m0 = (const float*)d_in[2];
  const float* ie = (const float*)d_in[3];
  const float* he = (const float*)d_in[4];
  const float* me = (const float*)d_in[5];
  const float* wi = (const float*)d_in[6];
  const float* wh = (const float*)d_in[7];
  const float* wm = (const float*)d_in[8];
  const float* at = (const float*)d_in[9];
  const float* bt = (const float*)d_in[10];
  float* out = (float*)d_out;

  char* ws = (char*)d_ws;
  unsigned* counter = (unsigned*)ws;                         // 256 B
  float*  atf = (float*)(ws + 256);                          // 256 KB
  __bf16* wif = (__bf16*)(ws + 256 + 262144);                // 128 KB
  __bf16* whf = (__bf16*)(ws + 256 + 262144 + 131072);       // 512 KB
  __bf16* wmf = (__bf16*)(ws + 256 + 262144 + 131072 + 524288); // 256 KB

  hipMemsetAsync(counter, 0, 256, stream);
  lmu_pack_at<<<256, 256, 0, stream>>>(at, atf);             // 65536 elems
  lmu_pack_w<<<256, 256, 0, stream>>>(wi, wif, LMU_D);       // 128*512
  lmu_pack_w<<<1024, 256, 0, stream>>>(wh, whf, LMU_U);      // 512*512
  lmu_pack_w<<<512, 256, 0, stream>>>(wm, wmf, LMU_N);       // 256*512
  lmu_scan<<<LMU_G, 256, 0, stream>>>(x, h0, m0, ie, he, me, bt, atf, wif,
                                      whf, wmf, out, counter);
}
